// Qwen3_5Attention_60026462929047
// MI455X (gfx1250) — compile-verified, matrix operand order checked by reading, not run on hardware
//
#include <hip/hip_runtime.h>
#include <hip/hip_bf16.h>

// ---------------------------------------------------------------------------
// Problem constants (from the reference)
// ---------------------------------------------------------------------------
constexpr int CB   = 2;
constexpr int CS   = 2048;
constexpr int CHID = 2048;
constexpr int CH   = 16;
constexpr int CKV  = 4;
constexpr int CD   = 128;
constexpr int CROT = 64;
constexpr int CEMB = 128;
constexpr int NTOT = CH * 2 * CD + CKV * CD + CKV * CD + CEMB; // 5248
constexpr int CM   = CB * CS;                                  // 4096 rows

typedef __attribute__((ext_vector_type(16))) __bf16 v16bf;
typedef __attribute__((ext_vector_type(8)))  __bf16 v8bf;
typedef __attribute__((ext_vector_type(8)))  float  v8f;

// ---------------------------------------------------------------------------
// WMMA fragment load: 16x32 bf16 tile from a row-major matrix (stride elems).
// Matches CDNA5 16-bit A/B layout: lanes 0-15 -> row r, K = {0..7, 16..23};
// lanes 16-31 -> row r, K = {8..15, 24..31}.  Two b128 loads per lane.
// ---------------------------------------------------------------------------
__device__ __forceinline__ v16bf load_frag(const __bf16* tile, int stride) {
  int lane = threadIdx.x & 31;
  int r    = lane & 15;
  int ko   = (lane & 16) ? 8 : 0;
  const v8bf* p0 = (const v8bf*)(tile + (size_t)r * stride + ko);
  const v8bf* p1 = (const v8bf*)(tile + (size_t)r * stride + ko + 16);
  v8bf lo = *p0;
  v8bf hi = *p1;
  return __builtin_shufflevector(lo, hi, 0,1,2,3,4,5,6,7,8,9,10,11,12,13,14,15);
}

__device__ __forceinline__ v8f wmma_bf16(v16bf a, v16bf b, v8f c) {
  return __builtin_amdgcn_wmma_f32_16x16x32_bf16(false, a, false, b, (short)0, c,
                                                 false, false);
}

// Async global -> LDS copy of one 16-byte chunk (ASYNCcnt tracked).
// The offset: field shifts both the LDS and the global address.
__device__ __forceinline__ void async_copy_b128(unsigned lds_addr,
                                                const void* gptr) {
  unsigned long long ga = (unsigned long long)gptr;
  asm volatile("global_load_async_to_lds_b128 %0, %1, off"
               :: "v"(lds_addr), "v"(ga) : "memory");
}
__device__ __forceinline__ void async_copy_b128_off16(unsigned lds_addr,
                                                      const void* gptr) {
  unsigned long long ga = (unsigned long long)gptr;
  asm volatile("global_load_async_to_lds_b128 %0, %1, off offset:16"
               :: "v"(lds_addr), "v"(ga) : "memory");
}

// ---------------------------------------------------------------------------
// Elementwise conversion / packing kernels
// ---------------------------------------------------------------------------
__global__ void f32_to_bf16_kernel(const float* __restrict__ src,
                                   __bf16* __restrict__ dst, int n) {
  int i = blockIdx.x * 256 + threadIdx.x;
  if (i < n) dst[i] = (__bf16)src[i];
}

// Bt[(colOff + n)*ldBt + k] = W[k*N + n]   (weight transpose + bf16 convert)
__global__ void pack_wt_kernel(const float* __restrict__ W, __bf16* __restrict__ Bt,
                               int K, int N, int ldBt, int colOff) {
  int i = blockIdx.x * 256 + threadIdx.x;
  if (i >= K * N) return;
  int k = i / N;
  int n = i % N;
  Bt[(size_t)(colOff + n) * ldBt + k] = (__bf16)W[(size_t)k * N + n];
}

// ---------------------------------------------------------------------------
// bf16 WMMA GEMM:  C[M,N] = A[M,K] * Bt[N,K]^T  (f32 accumulate)
// block = 128 threads (4 waves); block tile 64x64; wave tile 16x64.
// The 64x32 B tile (shared by all 4 waves) is staged through LDS with
// double-buffered global_load_async_to_lds_b128; A fragments come straight
// from global (prefetched). Requires M%64==0, N%64==0, K%32==0, K>=64.
// ---------------------------------------------------------------------------
__global__ void gemm_bf16_kernel(const __bf16* __restrict__ A,
                                 const __bf16* __restrict__ Bt,
                                 float* __restrict__ C, int M, int N, int K) {
  __shared__ __attribute__((aligned(16))) __bf16 Bsh[2][64 * 32]; // 2 x 4KB

  int tid  = threadIdx.x;
  int wave = tid >> 5;
  int lane = tid & 31;
  int row0 = blockIdx.y * 64 + wave * 16;
  int col0 = blockIdx.x * 64;

  // Async-copy addressing: thread t moves column n = t/2, k-span (t&1)*16..+15
  // of the current 64x32 tile (two b128 chunks via offset:16).
  int nIdx = tid >> 1;
  int kOff = (tid & 1) * 16;
  const __bf16* gB = Bt + (size_t)(col0 + nIdx) * K + kOff;
  unsigned lbase0 = (unsigned)(uintptr_t)&Bsh[0][0];
  unsigned lbase1 = (unsigned)(uintptr_t)&Bsh[1][0];
  unsigned loff   = (unsigned)(nIdx * 64 + kOff * 2); // bytes, [64][32] row-major

  v8f vz = {};
  v8f acc[4];
#pragma unroll
  for (int t = 0; t < 4; ++t) acc[t] = vz;

  // Prologue: fill buffer 0 with the first tile.
  async_copy_b128(lbase0 + loff, gB);
  async_copy_b128_off16(lbase0 + loff, gB);

  int pb = 0;
  for (int k0 = 0; k0 < K; k0 += 32) {
    if (k0 + 32 < K) {
      unsigned l = (pb ? lbase0 : lbase1) + loff;
      const __bf16* g = gB + k0 + 32;
      async_copy_b128(l, g);
      async_copy_b128_off16(l, g);
      // 2 newest ops (next tile) may remain in flight; current tile is done.
      asm volatile("s_wait_asynccnt 0x2" ::: "memory");
    } else {
      asm volatile("s_wait_asynccnt 0x0" ::: "memory");
    }
    __syncthreads(); // all waves' async chunks visible

    const __bf16* Bbuf = pb ? &Bsh[1][0] : &Bsh[0][0];
    __builtin_prefetch(A + (size_t)row0 * K + k0 + 64, 0, 1); // pull A stream
    v16bf a = load_frag(A + (size_t)row0 * K + k0, K);
#pragma unroll
    for (int t = 0; t < 4; ++t) {
      v16bf b = load_frag(Bbuf + t * 16 * 32, 32); // LDS ds_load_b128 x2
      acc[t] = wmma_bf16(a, b, acc[t]);
    }
    __syncthreads(); // done reading before buffer is overwritten again
    pb ^= 1;
  }

  int cbase = (lane & 16) ? 8 : 0;
  int ccol  = lane & 15;
#pragma unroll
  for (int t = 0; t < 4; ++t)
#pragma unroll
    for (int i = 0; i < 8; ++i)
      C[(size_t)(row0 + cbase + i) * N + col0 + t * 16 + ccol] = acc[t][i];
}

// ---------------------------------------------------------------------------
// Token post-processing: layernorm + gelu + Wd + gumbel mask; per-head
// RMSNorm + mask + RoPE; emits Q/K bf16 [*,S,D] and V transposed [*,D,S].
// block = 128 threads, grid = CM tokens.
// ---------------------------------------------------------------------------
__device__ __forceinline__ float block_reduce_sum(float v, float* buf) {
  int t = threadIdx.x;
  buf[t] = v;
  __syncthreads();
  for (int o = 64; o > 0; o >>= 1) {
    if (t < o) buf[t] += buf[t + o];
    __syncthreads();
  }
  float r = buf[0];
  __syncthreads();
  return r;
}

__device__ __forceinline__ float hash_uniform(unsigned x) {
  x ^= x >> 17; x *= 0xed5ad4bbu;
  x ^= x >> 11; x *= 0xac4c1b51u;
  x ^= x >> 15; x *= 0x31848babu;
  x ^= x >> 14;
  return ((float)(x >> 8) + 0.5f) * (1.0f / 16777216.0f);
}

__global__ void token_post_kernel(const float* __restrict__ Call,
                                  const float* __restrict__ cosT,
                                  const float* __restrict__ sinT,
                                  const float* __restrict__ Wd,
                                  const float* __restrict__ ln_g,
                                  const float* __restrict__ ln_b,
                                  const float* __restrict__ rnn,
                                  const float* __restrict__ qnw,
                                  const float* __restrict__ knw,
                                  __bf16* __restrict__ Qbuf,
                                  __bf16* __restrict__ Kbuf,
                                  __bf16* __restrict__ Vt) {
  __shared__ float sh[128];
  __shared__ float sred[128];
  __shared__ float smask[128];

  int m = blockIdx.x;
  int b = m / CS;
  int s = m % CS;
  int t = threadIdx.x;
  const float* row = Call + (size_t)m * NTOT;

  // ---- gating mask path: layernorm(emb+rnn) -> gelu -> @Wd -> gumbel ----
  float e  = row[CH * 2 * CD + 2 * CKV * CD + t] + rnn[t];
  float mu = block_reduce_sum(e, sred) * (1.0f / 128.0f);
  float d  = e - mu;
  float var = block_reduce_sum(d * d, sred) * (1.0f / 128.0f);
  float ln  = d * rsqrtf(var + 1e-5f) * ln_g[t] + ln_b[t];
  float gl  = 0.5f * ln * (1.0f + erff(ln * 0.70710678118f));
  sh[t] = gl;
  __syncthreads();

  float logit = 0.f;
#pragma unroll 4
  for (int e2 = 0; e2 < 128; ++e2) logit += sh[e2] * Wd[e2 * 128 + t];

  float u   = hash_uniform(0x9E3779B9u ^ (unsigned)(m * 128 + t));
  float gum = -logf(-logf(u));                       // -log(Exp(1))
  float y   = 1.f / (1.f + __expf(-((logit + gum + 3.0f) * 2.5f))); // /TAU=0.4
  smask[t]  = rintf(y);                              // straight-through round
  __syncthreads();

  float cv = (t < CROT) ? cosT[(size_t)m * CROT + t] : 0.f;
  float sv = (t < CROT) ? sinT[(size_t)m * CROT + t] : 0.f;

  // ---- q heads: rmsnorm -> mask -> rope ----
  for (int h = 0; h < CH; ++h) {
    float q  = row[h * 2 * CD + t];
    float ss = block_reduce_sum(q * q, sred) * (1.0f / 128.0f);
    q = q * rsqrtf(ss + 1e-6f) * (1.f + qnw[t]);
    q *= smask[t];
    sh[t] = q;
    __syncthreads();
    float o = q;
    if (t < CROT) {
      float other = (t < 32) ? -sh[t + 32] : sh[t - 32];
      o = sh[t] * cv + other * sv;
    }
    Qbuf[((size_t)(b * CH + h) * CS + s) * CD + t] = (__bf16)o;
    __syncthreads();
  }

  // ---- k heads + v ----
  for (int kv = 0; kv < CKV; ++kv) {
    float k  = row[CH * 2 * CD + kv * CD + t];
    float ss = block_reduce_sum(k * k, sred) * (1.0f / 128.0f);
    k = k * rsqrtf(ss + 1e-6f) * (1.f + knw[t]);
    k *= smask[t];
    sh[t] = k;
    __syncthreads();
    float o = k;
    if (t < CROT) {
      float other = (t < 32) ? -sh[t + 32] : sh[t - 32];
      o = sh[t] * cv + other * sv;
    }
    Kbuf[((size_t)(b * CKV + kv) * CS + s) * CD + t] = (__bf16)o;
    __syncthreads();

    float vv = row[CH * 2 * CD + CKV * CD + kv * CD + t];
    Vt[((size_t)(b * CKV + kv) * CD + t) * CS + s] = (__bf16)vv;  // transposed
  }
}

// ---------------------------------------------------------------------------
// Flash attention (causal, GQA rep=4) with WMMA. block = 256 (8 waves);
// each wave owns a 16-row query tile; keys processed in 64-wide chunks.
// P staged through LDS (bf16) to re-layout C-tile -> A-fragment.
// Gate (sigmoid) applied in epilogue; output bf16 [B*S, H*D].
// ---------------------------------------------------------------------------
__global__ void attn_fwd_kernel(const __bf16* __restrict__ Qbuf,
                                const __bf16* __restrict__ Kbuf,
                                const __bf16* __restrict__ Vt,
                                const float* __restrict__ Call,
                                __bf16* __restrict__ Obuf) {
  __shared__ __attribute__((aligned(16))) __bf16 Psh[8][16 * 64];

  int bh   = blockIdx.y;
  int b    = bh / CH;
  int h    = bh % CH;
  int kv   = h / (CH / CKV);
  int wave = threadIdx.x >> 5;
  int lane = threadIdx.x & 31;
  int q0   = blockIdx.x * 128 + wave * 16;

  const __bf16* Q  = Qbuf + (size_t)(b * CH + h) * CS * CD;   // [S][D]
  const __bf16* Km = Kbuf + (size_t)(b * CKV + kv) * CS * CD; // [S][D]
  const __bf16* V  = Vt + (size_t)(b * CKV + kv) * CD * CS;   // [D][S]
  __bf16* P = &Psh[wave][0];

  int cbase = (lane & 16) ? 8 : 0;
  int ccol  = lane & 15;

  float mrow[8], lrow[8];
  v8f O[8];
  v8f vz = {};
#pragma unroll
  for (int i = 0; i < 8; ++i) { mrow[i] = -1e30f; lrow[i] = 0.f; O[i] = vz; }

  const float scal = 0.08838834764831845f; // 1/sqrt(128)

  for (int kb = 0; kb < q0 + 16; kb += 64) {
    // ---- S = Q K^T over this 64-key chunk ----
    v8f Sacc[4];
#pragma unroll
    for (int t4 = 0; t4 < 4; ++t4) Sacc[t4] = vz;
    for (int d0 = 0; d0 < CD; d0 += 32) {
      v16bf a = load_frag(Q + (size_t)q0 * CD + d0, CD);
#pragma unroll
      for (int t4 = 0; t4 < 4; ++t4) {
        v16bf bb = load_frag(Km + (size_t)(kb + t4 * 16) * CD + d0, CD);
        Sacc[t4] = wmma_bf16(a, bb, Sacc[t4]);
      }
    }

    // ---- causal mask + online softmax (row stats via half-wave shuffles) ----
#pragma unroll
    for (int i = 0; i < 8; ++i) {
      int rr = q0 + cbase + i;
      float mx = -1e30f;
#pragma unroll
      for (int t4 = 0; t4 < 4; ++t4) {
        float v0 = Sacc[t4][i] * scal;
        int key = kb + t4 * 16 + ccol;
        if (key > rr) v0 = -1e30f;
        Sacc[t4][i] = v0;
        mx = fmaxf(mx, v0);
      }
#pragma unroll
      for (int off = 1; off < 16; off <<= 1)
        mx = fmaxf(mx, __shfl_xor(mx, off, 32));
      float nm = fmaxf(mrow[i], mx);
      float rescale = __expf(mrow[i] - nm);
      lrow[i] *= rescale;
#pragma unroll
      for (int dt = 0; dt < 8; ++dt) O[dt][i] *= rescale;
      float rs = 0.f;
#pragma unroll
      for (int t4 = 0; t4 < 4; ++t4) {
        float p = __expf(Sacc[t4][i] - nm);
        Sacc[t4][i] = p;
        rs += p;
      }
#pragma unroll
      for (int off = 1; off < 16; off <<= 1) rs += __shfl_xor(rs, off, 32);
      lrow[i] += rs;
      mrow[i] = nm;
    }

    // ---- stage P (C-layout -> [row][key] bf16 in LDS) ----
#pragma unroll
    for (int t4 = 0; t4 < 4; ++t4)
#pragma unroll
      for (int i = 0; i < 8; ++i)
        P[(cbase + i) * 64 + t4 * 16 + ccol] = (__bf16)Sacc[t4][i];
    asm volatile("s_wait_dscnt 0x0" ::: "memory");

    // ---- O += P V ----
#pragma unroll
    for (int kk = 0; kk < 2; ++kk) {
      v16bf pa = load_frag(P + kk * 32, 64);
#pragma unroll
      for (int dt = 0; dt < 8; ++dt) {
        v16bf vb = load_frag(V + (size_t)(dt * 16) * CS + kb + kk * 32, CS);
        O[dt] = wmma_bf16(pa, vb, O[dt]);
      }
    }
  }

  // ---- epilogue: normalize, gate, store bf16 [B*S, H*D] ----
#pragma unroll
  for (int dt = 0; dt < 8; ++dt)
#pragma unroll
    for (int i = 0; i < 8; ++i) {
      int rr = q0 + cbase + i;
      float o = O[dt][i] / lrow[i];
      float gt = Call[(size_t)(b * CS + rr) * NTOT + h * 2 * CD + CD + dt * 16 + ccol];
      o *= 1.f / (1.f + __expf(-gt));
      Obuf[(size_t)(b * CS + rr) * (CH * CD) + (size_t)h * CD + dt * 16 + ccol] =
          (__bf16)o;
    }
}

// ---------------------------------------------------------------------------
// Host-side orchestration
// ---------------------------------------------------------------------------
extern "C" void kernel_launch(void* const* d_in, const int* in_sizes, int n_in,
                              void* d_out, int out_size, void* d_ws, size_t ws_size,
                              hipStream_t stream) {
  (void)in_sizes; (void)n_in; (void)out_size; (void)ws_size;

  const float* hidden = (const float*)d_in[0];
  const float* cosT   = (const float*)d_in[1];
  const float* sinT   = (const float*)d_in[2];
  const float* Wq     = (const float*)d_in[3];
  const float* Wk     = (const float*)d_in[4];
  const float* Wv     = (const float*)d_in[5];
  const float* Wo     = (const float*)d_in[6];
  const float* qnw    = (const float*)d_in[7];
  const float* knw    = (const float*)d_in[8];
  const float* Wr     = (const float*)d_in[9];
  const float* Wd     = (const float*)d_in[10];
  const float* ln_g   = (const float*)d_in[11];
  const float* ln_b   = (const float*)d_in[12];
  const float* rnn    = (const float*)d_in[13];
  float* out = (float*)d_out;

  char* ws = (char*)d_ws;
  size_t off = 0;
  auto alloc = [&](size_t bytes) -> void* {
    void* p = ws + off;
    off += (bytes + 255) & ~(size_t)255;
    return p;
  };

  __bf16* A16   = (__bf16*)alloc((size_t)CM * CHID * 2);        // hidden bf16
  __bf16* BtAll = (__bf16*)alloc((size_t)NTOT * CHID * 2);      // packed W^T
  float*  Call  = (float*) alloc((size_t)CM * NTOT * 4);        // projections
  __bf16* Qb    = (__bf16*)alloc((size_t)CB * CH * CS * CD * 2);
  __bf16* Kb    = (__bf16*)alloc((size_t)CB * CKV * CS * CD * 2);
  __bf16* Vt    = (__bf16*)alloc((size_t)CB * CKV * CD * CS * 2);
  __bf16* Ob    = (__bf16*)alloc((size_t)CM * (CH * CD) * 2);
  __bf16* BtWo  = (__bf16*)alloc((size_t)CHID * (CH * CD) * 2);

  // 1) hidden -> bf16
  {
    int n = CM * CHID;
    f32_to_bf16_kernel<<<(n + 255) / 256, 256, 0, stream>>>(hidden, A16, n);
  }
  // 2) pack weights (transpose + bf16)
  {
    int n;
    n = CHID * (CH * 2 * CD);
    pack_wt_kernel<<<(n + 255) / 256, 256, 0, stream>>>(Wq, BtAll, CHID, CH * 2 * CD, CHID, 0);
    n = CHID * (CKV * CD);
    pack_wt_kernel<<<(n + 255) / 256, 256, 0, stream>>>(Wk, BtAll, CHID, CKV * CD, CHID, CH * 2 * CD);
    pack_wt_kernel<<<(n + 255) / 256, 256, 0, stream>>>(Wv, BtAll, CHID, CKV * CD, CHID, CH * 2 * CD + CKV * CD);
    n = CHID * CEMB;
    pack_wt_kernel<<<(n + 255) / 256, 256, 0, stream>>>(Wr, BtAll, CHID, CEMB, CHID, CH * 2 * CD + 2 * CKV * CD);
    n = (CH * CD) * CHID;
    pack_wt_kernel<<<(n + 255) / 256, 256, 0, stream>>>(Wo, BtWo, CH * CD, CHID, CH * CD, 0);
  }
  // 3) fused projection GEMM: [4096 x 5248] = A16 * BtAll^T
  {
    dim3 grid(NTOT / 64, CM / 64);
    gemm_bf16_kernel<<<grid, 128, 0, stream>>>(A16, BtAll, Call, CM, NTOT, CHID);
  }
  // 4) per-token post-processing (mask/rmsnorm/rope, Q/K/V emit)
  token_post_kernel<<<CM, 128, 0, stream>>>(Call, cosT, sinT, Wd, ln_g, ln_b, rnn,
                                            qnw, knw, Qb, Kb, Vt);
  // 5) flash attention
  {
    dim3 grid(CS / 128, CB * CH);
    attn_fwd_kernel<<<grid, 256, 0, stream>>>(Qb, Kb, Vt, Call, Ob);
  }
  // 6) output GEMM: out[4096 x 2048] = Ob * BtWo^T
  {
    dim3 grid(CHID / 64, CM / 64);
    gemm_bf16_kernel<<<grid, 128, 0, stream>>>(Ob, BtWo, out, CM, CHID, CH * CD);
  }
}
